// ESN_21208548508059
// MI455X (gfx1250) — matrix operand order repaired and checked by reading, not compiled, not run on hardware
//
#include <hip/hip_runtime.h>
#include <hip/hip_bf16.h>

#define T_STEPS 16384
#define R_DIM   1024
#define I_DIM   128
#define NBLK    128            // persistent scan blocks
#define ROWS_PB (R_DIM / NBLK) // 8 rows of W_res per block

typedef __attribute__((ext_vector_type(16))) __bf16 v16bf;
typedef __attribute__((ext_vector_type(8)))  float  v8f;

// ---------------------------------------------------------------------------
// Split fp32 -> (hi, lo) bf16 pair:  x ~= hi + lo, each rounded RNE.
// ---------------------------------------------------------------------------
__global__ void cvt_split_bf16(const float* __restrict__ src,
                               __bf16* __restrict__ hi,
                               __bf16* __restrict__ lo, int n) {
    int i = blockIdx.x * blockDim.x + threadIdx.x;
    if (i < n) {
        float x = src[i];
        __bf16 h = (__bf16)x;
        float r = x - (float)h;
        hi[i] = h;
        lo[i] = (__bf16)r;
    }
}

// ---------------------------------------------------------------------------
// Phase 1: U = X @ W_in^T   (T x R, K = 128), bf16 WMMA with 3-product split.
// One 16x16 output tile per wave; 8 waves per block.
// ---------------------------------------------------------------------------
__global__ void __launch_bounds__(256)
esn_input_gemm(const __bf16* __restrict__ Xhi, const __bf16* __restrict__ Xlo,
               const __bf16* __restrict__ Whi, const __bf16* __restrict__ Wlo,
               float* __restrict__ U) {
    const int wave = threadIdx.x >> 5;
    const int lane = threadIdx.x & 31;
    const int lo16 = lane & 15;
    const int hi16 = lane >> 4;

    const int tile    = blockIdx.x * 8 + wave;       // 65536 tiles total
    const int rowTile = tile >> 6;                   // T/16 = 1024
    const int colTile = tile & 63;                   // R/16 = 64

    const int arow = rowTile * 16 + lo16;            // A: M = lane&15
    const int bcol = colTile * 16 + lo16;            // B: N = lane&15

    union Frag { uint4 q[2]; v16bf v; };

    v8f acc = {};

    #pragma unroll
    for (int k0 = 0; k0 < I_DIM; k0 += 32) {
        // A fragment: lane holds K = k0 + hi16*8 + [0..7]  and  +16 more.
        Frag ah, al, bh, bl;
        const __bf16* ap_h = Xhi + (size_t)arow * I_DIM + k0 + hi16 * 8;
        const __bf16* ap_l = Xlo + (size_t)arow * I_DIM + k0 + hi16 * 8;
        ah.q[0] = *(const uint4*)(ap_h);
        ah.q[1] = *(const uint4*)(ap_h + 16);
        al.q[0] = *(const uint4*)(ap_l);
        al.q[1] = *(const uint4*)(ap_l + 16);

        // B fragment: lane holds column n = lane&15, K = k0 + hi16*8 + [0..7]  and  +16 more.
        const __bf16* bp_h = Whi + (size_t)bcol * I_DIM + k0 + hi16 * 8;
        const __bf16* bp_l = Wlo + (size_t)bcol * I_DIM + k0 + hi16 * 8;
        bh.q[0] = *(const uint4*)(bp_h);
        bh.q[1] = *(const uint4*)(bp_h + 16);
        bl.q[0] = *(const uint4*)(bp_l);
        bl.q[1] = *(const uint4*)(bp_l + 16);

        // 3-product split: hi*hi + hi*lo + lo*hi  (~fp32 accuracy)
        acc = __builtin_amdgcn_wmma_f32_16x16x32_bf16(false, ah.v, false, bh.v,
                                                      (short)0, acc, false, false);
        acc = __builtin_amdgcn_wmma_f32_16x16x32_bf16(false, ah.v, false, bl.v,
                                                      (short)0, acc, false, false);
        acc = __builtin_amdgcn_wmma_f32_16x16x32_bf16(false, al.v, false, bh.v,
                                                      (short)0, acc, false, false);
    }

    // C/D layout: VGPR v -> M = v + 8*hi16, N = lane&15
    #pragma unroll
    for (int v = 0; v < 8; ++v) {
        int m = rowTile * 16 + v + hi16 * 8;
        int n = colTile * 16 + lo16;
        U[(size_t)m * R_DIM + n] = acc[v];
    }
}

// ---------------------------------------------------------------------------
// Sync counter reset (every launch -> graph-replay deterministic).
// ---------------------------------------------------------------------------
__global__ void esn_init(int* sync) {
    if (threadIdx.x == 0) *sync = 0;
}

// ---------------------------------------------------------------------------
// Phase 2: persistent scan.  Block b owns rows [b*8, b*8+8) of W_res in LDS.
// Step t: stage s_{t-1} (read from out[t-1], or state0 at t=0) into LDS,
// one wave per row does the fp32 dot, tanh, write out[t], grid barrier.
// ---------------------------------------------------------------------------
__global__ void __launch_bounds__(256)
esn_scan(const float* __restrict__ Wres, const float* __restrict__ U,
         const float* __restrict__ s0, float* out, int* sync) {
    __shared__ float Wlds[ROWS_PB * R_DIM];   // 32 KB
    __shared__ float s_sh[R_DIM];             //  4 KB

    const int tid  = threadIdx.x;
    const int r0   = blockIdx.x * ROWS_PB;
    const int jrow = tid >> 5;                // wave id = row within slice (0..7)
    const int kl   = tid & 31;                // lane within wave

    // Rows r0..r0+7 are contiguous: straight float4 copy into LDS.
    {
        const float4* src = (const float4*)(Wres + (size_t)r0 * R_DIM);
        float4*       dst = (float4*)Wlds;
        for (int i = tid; i < (ROWS_PB * R_DIM) / 4; i += 256) dst[i] = src[i];
    }
    __syncthreads();

    for (int t = 0; t < T_STEPS; ++t) {
        const float* sprev = (t == 0) ? s0 : (out + (size_t)(t - 1) * R_DIM);
        // stage full state into LDS (256 float4 = 1024 floats)
        ((float4*)s_sh)[tid] = ((const float4*)sprev)[tid];
        __syncthreads();

        // fp32 dot: 32 lanes x 8 float4 chunks = 1024 MACs per row
        const float4* w4 = (const float4*)(Wlds + jrow * R_DIM);
        const float4* s4 = (const float4*)s_sh;
        float acc = 0.0f;
        #pragma unroll
        for (int c = 0; c < R_DIM / 4; c += 32) {
            float4 w = w4[c + kl];
            float4 s = s4[c + kl];
            acc += w.x * s.x + w.y * s.y + w.z * s.z + w.w * s.w;
        }
        acc += __shfl_xor(acc, 16, 32);
        acc += __shfl_xor(acc, 8, 32);
        acc += __shfl_xor(acc, 4, 32);
        acc += __shfl_xor(acc, 2, 32);
        acc += __shfl_xor(acc, 1, 32);

        if (kl == 0) {
            int r = r0 + jrow;
            float y = tanhf(U[(size_t)t * R_DIM + r] + acc);
            out[(size_t)t * R_DIM + r] = y;
        }

        // ---- device-wide barrier (monotonic counter, agent scope) ----
        __threadfence();          // publish out[t] rows
        __syncthreads();          // also separates s_sh readers from restage
        if (tid == 0) {
            __hip_atomic_fetch_add(sync, 1, __ATOMIC_RELEASE,
                                   __HIP_MEMORY_SCOPE_AGENT);
            const int target = NBLK * (t + 1);
            while (__hip_atomic_load(sync, __ATOMIC_ACQUIRE,
                                     __HIP_MEMORY_SCOPE_AGENT) < target) {
                __builtin_amdgcn_s_sleep(1);
            }
        }
        __syncthreads();
        __threadfence();          // acquire side for all lanes
    }
}

// ---------------------------------------------------------------------------
extern "C" void kernel_launch(void* const* d_in, const int* in_sizes, int n_in,
                              void* d_out, int out_size, void* d_ws, size_t ws_size,
                              hipStream_t stream) {
    const float* X     = (const float*)d_in[0];   // (T, I)
    const float* W_in  = (const float*)d_in[1];   // (R, I)
    const float* W_res = (const float*)d_in[2];   // (R, R)
    const float* s0    = (const float*)d_in[3];   // (R,)
    float* out = (float*)d_out;                   // (T, R)

    char* ws = (char*)d_ws;
    size_t off = 0;
    float* U = (float*)(ws + off);  off += (size_t)T_STEPS * R_DIM * sizeof(float);
    __bf16* Xhi = (__bf16*)(ws + off); off += (size_t)T_STEPS * I_DIM * sizeof(__bf16);
    __bf16* Xlo = (__bf16*)(ws + off); off += (size_t)T_STEPS * I_DIM * sizeof(__bf16);
    __bf16* Whi = (__bf16*)(ws + off); off += (size_t)R_DIM * I_DIM * sizeof(__bf16);
    __bf16* Wlo = (__bf16*)(ws + off); off += (size_t)R_DIM * I_DIM * sizeof(__bf16);
    off = (off + 127) & ~(size_t)127;
    int* sync = (int*)(ws + off);

    // bf16 hi/lo splits of X and W_in
    {
        int nX = T_STEPS * I_DIM;
        cvt_split_bf16<<<(nX + 255) / 256, 256, 0, stream>>>(X, Xhi, Xlo, nX);
        int nW = R_DIM * I_DIM;
        cvt_split_bf16<<<(nW + 255) / 256, 256, 0, stream>>>(W_in, Whi, Wlo, nW);
    }

    // Phase 1: U = X @ W_in^T via WMMA (65536 tiles / 8 waves per block)
    esn_input_gemm<<<(T_STEPS / 16) * (R_DIM / 16) / 8, 256, 0, stream>>>(
        Xhi, Xlo, Whi, Wlo, U);

    // Phase 2: persistent sequential scan
    esn_init<<<1, 64, 0, stream>>>(sync);
    esn_scan<<<NBLK, 256, 0, stream>>>(W_res, U, s0, out, sync);
}
